// sLSTMCell_41154376630679
// MI455X (gfx1250) — matrix operand order repaired and checked by reading, not compiled, not run on hardware
//
#include <hip/hip_runtime.h>
#include <hip/hip_bf16.h>

#define B_SZ 4096
#define D_SZ 2048
#define BK   32
#define NK   (D_SZ / BK)      // 64 K-steps
#define TROW 40               // padded LDS row stride (elements) -> bank-conflict free
#define TEL  (128 * TROW)     // one padded 128x32 tile = 5120 elements
#define NTILES 11             // X, U, H, 8 weights
#define BUF_EL (NTILES * TEL) // 56320 elements per buffer
#define SMEM_BYTES (2 * BUF_EL * 2)  // double-buffered, bf16 -> 225280 B (< 320KB WGP LDS)

typedef __attribute__((ext_vector_type(8)))  float  v8f;
typedef __attribute__((ext_vector_type(8)))  __bf16 v8bf;
typedef __attribute__((ext_vector_type(16))) __bf16 v16bf;
typedef __attribute__((ext_vector_type(4)))  int    i4v;

// ---- gfx1250 async global->LDS path (ASYNCcnt), with safe fallback --------
#if defined(__has_builtin)
#if __has_builtin(__builtin_amdgcn_global_load_async_to_lds_b128) && \
    __has_builtin(__builtin_amdgcn_s_wait_asynccnt)
#define USE_ASYNC_LDS 1
#endif
#endif

typedef __attribute__((address_space(1))) i4v gi4v;  // global int4
typedef __attribute__((address_space(3))) i4v si4v;  // LDS int4

__device__ __forceinline__ void cp_b128(__bf16* dst_lds, const __bf16* src_g) {
#ifdef USE_ASYNC_LDS
  __builtin_amdgcn_global_load_async_to_lds_b128(
      (gi4v*)(void*)src_g, (si4v*)(void*)dst_lds, 0, 0);
#else
  *(uint4*)dst_lds = *(const uint4*)src_g;
#endif
}

__device__ __forceinline__ void wait_async_cp() {
#ifdef USE_ASYNC_LDS
  __builtin_amdgcn_s_wait_asynccnt(0);
#endif
}

// ---------------------------------------------------------------------------
// Prep: u = silu(conv1d_same(x)); bf16 conversions (pad_low=1, pad_high=2)
// ---------------------------------------------------------------------------
__global__ __launch_bounds__(256) void prep_x_kernel(
    const float* __restrict__ x, const float* __restrict__ ck,
    const float* __restrict__ cb, __bf16* __restrict__ xbf,
    __bf16* __restrict__ ubf) {
  int i = blockIdx.x * 256 + threadIdx.x;
  int d = i & (D_SZ - 1);
  float xm1 = (d >= 1) ? x[i - 1] : 0.0f;
  float x0  = x[i];
  float xp1 = (d <= D_SZ - 2) ? x[i + 1] : 0.0f;
  float xp2 = (d <= D_SZ - 3) ? x[i + 2] : 0.0f;
  float u = ck[0] * xm1 + ck[1] * x0 + ck[2] * xp1 + ck[3] * xp2 + cb[0];
  float su = u / (1.0f + __expf(-u));  // SiLU
  xbf[i] = (__bf16)x0;
  ubf[i] = (__bf16)su;
}

__global__ __launch_bounds__(256) void prep_h_kernel(
    const float* __restrict__ h, __bf16* __restrict__ hbf) {
  int i = blockIdx.x * 256 + threadIdx.x;
  hbf[i] = (__bf16)h[i];
}

// LDS-tiled transpose+convert: dst[n*D + k] = bf16(src[k*D + n])
__global__ __launch_bounds__(256) void wtrans_kernel(
    const float* __restrict__ src, __bf16* __restrict__ dst) {
  __shared__ __bf16 t[32][33];
  const int k0 = blockIdx.x * 32;
  const int n0 = blockIdx.y * 32;
  const int tx = threadIdx.x;   // 0..31
  const int ty = threadIdx.y;   // 0..7
#pragma unroll
  for (int yy = ty; yy < 32; yy += 8)
    t[yy][tx] = (__bf16)src[(size_t)(k0 + yy) * D_SZ + n0 + tx];  // coalesced
  __syncthreads();
#pragma unroll
  for (int yy = ty; yy < 32; yy += 8)
    dst[(size_t)(n0 + yy) * D_SZ + k0 + tx] = t[tx][yy];          // coalesced
}

// ---------------------------------------------------------------------------
// Fragment helpers (wave32 WMMA bf16 16x16x32 layouts, padded LDS stride 40)
// ---------------------------------------------------------------------------
__device__ __forceinline__ v16bf frag_cat(v8bf lo, v8bf hi) {
  return __builtin_shufflevector(lo, hi, 0, 1, 2, 3, 4, 5, 6, 7, 8, 9, 10, 11,
                                 12, 13, 14, 15);
}

__device__ __forceinline__ v16bf load_a_frag(const __bf16* t, int row, int hs) {
  // lane<16: row=lane, K{0..7,16..23}; lane>=16: K{8..15,24..31}
  const __bf16* p = t + row * TROW + hs * 8;
  return frag_cat(*(const v8bf*)p, *(const v8bf*)(p + 16));
}

__device__ __forceinline__ v16bf load_b_frag(const __bf16* t, int col, int hs) {
  // lane<16: col=lane, K{0..7,16..23}; lane>=16: K{8..15,24..31} ([n][k] tile)
  const __bf16* p = t + col * TROW + hs * 8;
  return frag_cat(*(const v8bf*)p, *(const v8bf*)(p + 16));
}

__device__ __forceinline__ v8f wmma_bf16(v16bf a, v16bf b, v8f c) {
  return __builtin_amdgcn_wmma_f32_16x16x32_bf16(false, a, false, b, (short)0,
                                                 c, false, false);
}

// Issue one K-step's tile fetches (11 x one b128 chunk per thread)
__device__ __forceinline__ void tile_loads(
    const __bf16* __restrict__ xbf, const __bf16* __restrict__ ubf,
    const __bf16* __restrict__ hbf, const __bf16* __restrict__ wts,
    int b0, int d0, int kb, __bf16* sbuf, int tid) {
  const int row = tid >> 2;          // 0..127
  const int colc = (tid & 3) * 8;    // 0,8,16,24
  const int lo = row * TROW + colc;  // LDS element offset within a tile
  const size_t DD = (size_t)D_SZ * D_SZ;
  const size_t aoff = (size_t)(b0 + row) * D_SZ + kb + colc;
  cp_b128(sbuf + 0 * TEL + lo, xbf + aoff);
  cp_b128(sbuf + 1 * TEL + lo, ubf + aoff);
  cp_b128(sbuf + 2 * TEL + lo, hbf + aoff);
  const size_t woff = (size_t)(d0 + row) * D_SZ + kb + colc;
#pragma unroll
  for (int m = 0; m < 8; ++m)
    cp_b128(sbuf + (3 + m) * TEL + lo, wts + (size_t)m * DD + woff);
}

// ---------------------------------------------------------------------------
// Fused sLSTM: 512 threads (16 waves), tile 128(b) x 128(d), BK=32.
// Wave w: column strip cs = w&7 (16 cols), row half rh = w>>3 (64 rows).
// Weight B-fragments loaded ONCE per K-step (8 regs), reused over 4 row
// subtiles x 4 gates x {W,R} = 32 WMMAs. Double-buffered LDS with async
// global->LDS prefetch of step kt+1 overlapping compute of step kt.
// ---------------------------------------------------------------------------
__global__ __launch_bounds__(512) void slstm_gemm_kernel(
    const __bf16* __restrict__ xbf, const __bf16* __restrict__ ubf,
    const __bf16* __restrict__ hbf, const __bf16* __restrict__ wts,
    const float* __restrict__ bz, const float* __restrict__ bi,
    const float* __restrict__ bfg, const float* __restrict__ bo,
    const float* __restrict__ cell, const float* __restrict__ norm,
    const float* __restrict__ stab, float* __restrict__ out) {
  // Weight order in wts (each D*D, [n][k]): 0 Wz 1 Rz 2 Wi 3 Ri 4 Wf 5 Rf 6 Wo 7 Ro
  extern __shared__ __align__(16) __bf16 smem[];

  const int tid = threadIdx.x;
  const int w = tid >> 5;
  const int lane = tid & 31;
  const int hs = lane >> 4;
  const int lrow = lane & 15;
  const int cs = w & 7;    // column strip (16 cols)
  const int rh = w >> 3;   // row half (64 rows)

  const int d0 = blockIdx.x * 128;
  const int b0 = blockIdx.y * 128;

  v8f acc_z[4], acc_i[4], acc_f[4], acc_o[4];
#pragma unroll
  for (int r = 0; r < 4; ++r) {
    acc_z[r] = (v8f)0.0f;
    acc_i[r] = (v8f)0.0f;
    acc_f[r] = (v8f)0.0f;
    acc_o[r] = (v8f)0.0f;
  }

  // Prologue: fetch tiles for kt=0 into buffer 0
  tile_loads(xbf, ubf, hbf, wts, b0, d0, 0, smem, tid);
  wait_async_cp();
  __syncthreads();

#pragma unroll 1
  for (int kt = 0; kt < NK; ++kt) {
    const __bf16* sb = smem + (kt & 1) * BUF_EL;
    __bf16* nb = smem + ((kt + 1) & 1) * BUF_EL;
    if (kt + 1 < NK)  // prefetch next K-step while computing this one
      tile_loads(xbf, ubf, hbf, wts, b0, d0, (kt + 1) * BK, nb, tid);

    // 8 weight fragments for this wave's column strip, loaded once
    v16bf bw[8];
#pragma unroll
    for (int m = 0; m < 8; ++m)
      bw[m] = load_b_frag(sb + (3 + m) * TEL, cs * 16 + lrow, hs);

#pragma unroll
    for (int r = 0; r < 4; ++r) {
      const int arow = rh * 64 + r * 16 + lrow;
      v16bf ax = load_a_frag(sb + 0 * TEL, arow, hs);
      v16bf au = load_a_frag(sb + 1 * TEL, arow, hs);
      v16bf ah = load_a_frag(sb + 2 * TEL, arow, hs);
      acc_z[r] = wmma_bf16(ax, bw[0], acc_z[r]);
      acc_z[r] = wmma_bf16(ah, bw[1], acc_z[r]);
      acc_i[r] = wmma_bf16(au, bw[2], acc_i[r]);
      acc_i[r] = wmma_bf16(ah, bw[3], acc_i[r]);
      acc_f[r] = wmma_bf16(au, bw[4], acc_f[r]);
      acc_f[r] = wmma_bf16(ah, bw[5], acc_f[r]);
      acc_o[r] = wmma_bf16(ax, bw[6], acc_o[r]);
      acc_o[r] = wmma_bf16(ah, bw[7], acc_o[r]);
    }
    if (kt + 1 < NK) wait_async_cp();
    __syncthreads();
  }

  // Epilogue: pointwise sLSTM update. C-layout: lanes 0-15 VGPR v -> M=v,
  // lanes 16-31 -> M=v+8; N = lane%16 within the wave's 16-col strip.
  const size_t BD = (size_t)B_SZ * D_SZ;
  const int d = d0 + cs * 16 + lrow;
  const float vbz = bz[d];
  const float vbi = bi[d];
  const float vbf = bfg[d];
  const float vbo = bo[d];
#pragma unroll
  for (int r = 0; r < 4; ++r) {
#pragma unroll
    for (int v = 0; v < 8; ++v) {
      const int b = b0 + rh * 64 + r * 16 + hs * 8 + v;
      const size_t idx = (size_t)b * D_SZ + d;
      const float zp = acc_z[r][v] + vbz;
      const float ip = acc_i[r][v] + vbi;
      const float fp = acc_f[r][v] + vbf;
      const float op = acc_o[r][v] + vbo;
      const float st = stab[idx];
      const float cs_ = cell[idx];
      const float ns = norm[idx];
      const float fps = fp + st;
      const float m_new = fmaxf(fps, ip);
      const float i_g = __expf(ip - m_new);
      const float f_g = __expf(fps - m_new);
      const float zt = tanhf(zp);
      const float og = 1.0f / (1.0f + __expf(-op));
      const float c_new = f_g * cs_ + i_g * zt;
      const float n_new = f_g * ns + i_g;
      const float h_new = og * c_new / n_new;
      out[idx] = c_new;
      out[BD + idx] = h_new;
      out[2 * BD + idx] = n_new;
      out[3 * BD + idx] = m_new;
    }
  }
}

// ---------------------------------------------------------------------------
extern "C" void kernel_launch(void* const* d_in, const int* in_sizes, int n_in,
                              void* d_out, int out_size, void* d_ws,
                              size_t ws_size, hipStream_t stream) {
  const float* x      = (const float*)d_in[0];
  const float* cell   = (const float*)d_in[1];
  const float* hidden = (const float*)d_in[2];
  const float* norm   = (const float*)d_in[3];
  const float* stab   = (const float*)d_in[4];
  const float* Wz = (const float*)d_in[5];
  const float* bz = (const float*)d_in[6];
  const float* Rz = (const float*)d_in[7];
  const float* Wi = (const float*)d_in[8];
  const float* bi = (const float*)d_in[9];
  const float* Ri = (const float*)d_in[10];
  const float* Wf = (const float*)d_in[11];
  const float* bf = (const float*)d_in[12];
  const float* Rf = (const float*)d_in[13];
  const float* Wo = (const float*)d_in[14];
  const float* bo = (const float*)d_in[15];
  const float* Ro = (const float*)d_in[16];
  const float* ck = (const float*)d_in[17];
  const float* cb = (const float*)d_in[18];

  const size_t BD = (size_t)B_SZ * D_SZ;
  const size_t DD = (size_t)D_SZ * D_SZ;
  __bf16* xbf = (__bf16*)d_ws;
  __bf16* ubf = xbf + BD;
  __bf16* hbf = ubf + BD;
  __bf16* wts = hbf + BD;  // 8 * DD bf16 (transposed [n][k])

  prep_x_kernel<<<BD / 256, 256, 0, stream>>>(x, ck, cb, xbf, ubf);
  prep_h_kernel<<<BD / 256, 256, 0, stream>>>(hidden, hbf);

  const float* wsrc[8] = {Wz, Rz, Wi, Ri, Wf, Rf, Wo, Ro};
  dim3 tgrid(D_SZ / 32, D_SZ / 32);
  dim3 tblk(32, 8);
  for (int g = 0; g < 8; ++g) {
    wtrans_kernel<<<tgrid, tblk, 0, stream>>>(wsrc[g], wts + (size_t)g * DD);
  }

  // Allow >64KB dynamic LDS (WGP supports 320KB); ignore failure.
  (void)hipFuncSetAttribute((const void*)slstm_gemm_kernel,
                            hipFuncAttributeMaxDynamicSharedMemorySize,
                            SMEM_BYTES);
  dim3 grid(D_SZ / 128, B_SZ / 128);
  slstm_gemm_kernel<<<grid, 512, SMEM_BYTES, stream>>>(
      xbf, ubf, hbf, wts, bz, bi, bf, bo, cell, norm, stab, (float*)d_out);
}
